// SelfAttention_17600775979527
// MI455X (gfx1250) — matrix operand order repaired and checked by reading, not compiled, not run on hardware
//
#include <hip/hip_runtime.h>
#include <math.h>

#define B_  8
#define S_  2048
#define DM_ 1024
#define DO_ 64

typedef __bf16 bf16;
typedef __attribute__((ext_vector_type(16))) __bf16 v16bf;
typedef __attribute__((ext_vector_type(8)))  __bf16 v8bf;
typedef __attribute__((ext_vector_type(2)))  __bf16 v2bf;
typedef __attribute__((ext_vector_type(8)))  float  v8f;

// workspace layout (bytes)
#define WT_BYTES   (3u * DO_ * DM_ * 2u)           // 3 transposed bf16 weights
#define PROJ_BYTES ((size_t)B_ * S_ * DO_ * 2u)    // one projected bf16 tensor

static __device__ __forceinline__ v8f wmma_bf16(v16bf a, v16bf b, v8f c) {
  // D = A(16x32 bf16) * B(32x16 bf16) + C(16x16 f32)
  return __builtin_amdgcn_wmma_f32_16x16x32_bf16(false, a, false, b, (short)0, c,
                                                 false, false);
}

// ---------------------------------------------------------------------------
// Kernel 0: Wt[mat][n][k] = (bf16) W[mat][k][n]   (n<64, k<1024)
// ---------------------------------------------------------------------------
__global__ void wt_transpose_kernel(const float* __restrict__ Wq,
                                    const float* __restrict__ Wk,
                                    const float* __restrict__ Wv,
                                    bf16* __restrict__ wt) {
  int idx = blockIdx.x * blockDim.x + threadIdx.x;
  if (idx >= 3 * DO_ * DM_) return;
  int mat = idx / (DO_ * DM_);
  int rem = idx % (DO_ * DM_);
  int n = rem / DM_;
  int k = rem % DM_;
  const float* W = (mat == 0) ? Wq : (mat == 1) ? Wk : Wv;
  wt[(size_t)mat * (DO_ * DM_) + (size_t)n * DM_ + k] = (bf16)W[(size_t)k * DO_ + n];
}

// ---------------------------------------------------------------------------
// Kernel 1: projection GEMM. One wave per block; block computes a 16x64 tile
// of X @ W + b (X is [B*S,1024] f32, result stored bf16).
// blockIdx.y selects q/k/v.
// ---------------------------------------------------------------------------
__global__ __launch_bounds__(32) void proj_kernel(
    const float* __restrict__ q, const float* __restrict__ k,
    const float* __restrict__ v, const float* __restrict__ bq,
    const float* __restrict__ bk, const float* __restrict__ bv,
    const bf16* __restrict__ wt, bf16* __restrict__ proj_base) {
  const int mat = blockIdx.y;
  const float* x    = (mat == 0) ? q  : (mat == 1) ? k  : v;
  const float* bias = (mat == 0) ? bq : (mat == 1) ? bk : bv;
  const bf16* Wt = wt + (size_t)mat * (DO_ * DM_);
  bf16* out = proj_base + (size_t)mat * ((size_t)B_ * S_ * DO_);

  const int row0  = blockIdx.x * 16;         // [0, B*S)
  const int lane  = threadIdx.x;
  const int half  = lane >> 4;               // 0: lanes 0-15, 1: lanes 16-31
  const int lm    = lane & 15;
  const int abase = half ? 16 : 0;           // A-layout K base within 32-chunk

  const float* xrow = x + (size_t)(row0 + lm) * DM_;

  v8f acc[4];
#pragma unroll
  for (int c = 0; c < 4; ++c)
#pragma unroll
    for (int r = 0; r < 8; ++r) acc[c][r] = 0.0f;

  for (int k0 = 0; k0 < DM_; k0 += 32) {
    // ---- A operand: 16x32 f32 tile row, packed to bf16, k = 16*half + s ----
    v16bf a;
    const float* p0 = xrow + k0 + abase;        // K = base + 0..7
    const float* p1 = xrow + k0 + abase + 8;    // K = base + 8..15
#pragma unroll
    for (int i = 0; i < 8; ++i) {
      a[i]     = (bf16)p0[i];
      a[8 + i] = (bf16)p1[i];
    }
    // ---- 4 N-chunks of 16 output features ----
#pragma unroll
    for (int c = 0; c < 4; ++c) {
      const bf16* bp = Wt + (size_t)(c * 16 + lm) * DM_ + k0 + (half ? 16 : 0);
      v16bf bm = *(const v16bf*)bp;             // 16 contiguous bf16 (32B aligned)
      acc[c] = wmma_bf16(a, bm, acc[c]);
    }
  }

#pragma unroll
  for (int c = 0; c < 4; ++c) {
    float bv_ = bias[c * 16 + lm];
#pragma unroll
    for (int r = 0; r < 8; ++r) {
      int row = row0 + r + 8 * half;            // D-tile: M = r + 8*(lane/16)
      out[(size_t)row * DO_ + c * 16 + lm] = (bf16)(acc[c][r] + bv_);
    }
  }
}

// ---------------------------------------------------------------------------
// Kernel 2: causal flash attention. One wave per block; each wave owns a
// 16-row query tile and streams 32-key tiles with online softmax.
// grid = (S/16, B)
// ---------------------------------------------------------------------------
__global__ __launch_bounds__(32) void attn_kernel(const bf16* __restrict__ qkv,
                                                  float* __restrict__ out) {
  const bf16* Qb = qkv;
  const bf16* Kb = qkv + (size_t)B_ * S_ * DO_;
  const bf16* Vb = qkv + 2 * (size_t)B_ * S_ * DO_;

  const int b     = blockIdx.y;
  const int qbase = blockIdx.x * 16;
  const int lane  = threadIdx.x;
  const int half  = lane >> 4;
  const int lm    = lane & 15;
  const int abase = half ? 16 : 0;

  __shared__ __align__(32) bf16 Pt[16 * 32];    // probs tile, [m][k]
  __shared__ __align__(32) bf16 Vt[64 * 32];    // V tile transposed, [n][k]

  // ---- preload Q tile (16x64) in A-layout, two K-chunks of 32 ----
  const bf16* Qrow = Qb + ((size_t)b * S_ + qbase + lm) * DO_;
  v16bf qa[2];
#pragma unroll
  for (int kc = 0; kc < 2; ++kc) {
    v8bf lo = *(const v8bf*)(Qrow + 32 * kc + abase);
    v8bf hi = *(const v8bf*)(Qrow + 32 * kc + abase + 8);
#pragma unroll
    for (int i = 0; i < 8; ++i) { qa[kc][i] = lo[i]; qa[kc][8 + i] = hi[i]; }
  }

  // ---- online-softmax state (per lane: rows r + 8*half) ----
  float M_[8], L_[8];
  v8f O[4];
#pragma unroll
  for (int r = 0; r < 8; ++r) { M_[r] = -__builtin_inff(); L_[r] = 0.0f; }
#pragma unroll
  for (int c = 0; c < 4; ++c)
#pragma unroll
    for (int r = 0; r < 8; ++r) O[c][r] = 0.0f;

  const int jend = qbase + 15;                  // causal bound (inclusive)
  for (int j0 = 0; j0 <= jend; j0 += 32) {
    // ---- prefetch next key tile (K and V rows) into caches ----
    int jn = j0 + 32;
    if (jn <= jend) {
      __builtin_prefetch(Kb + ((size_t)b * S_ + jn + lane) * DO_, 0, 3);
      __builtin_prefetch(Vb + ((size_t)b * S_ + jn + lane) * DO_, 0, 3);
    }

    __syncthreads();   // protect Vt/Pt reuse across iterations (WAR)

    // ---- stage V tile transposed into LDS, paired-key packing ----
    // lane = (half, t): rows j0+2t, j0+2t+1, features [32*half, 32*half+32).
    // Vt[n][2t],Vt[n][2t+1] are consecutive -> one b32 store per feature.
    {
      const int t    = lm;
      const int fsel = half * 32;
      const bf16* vr0 = Vb + ((size_t)b * S_ + j0 + 2 * t)     * DO_ + fsel;
      const bf16* vr1 = Vb + ((size_t)b * S_ + j0 + 2 * t + 1) * DO_ + fsel;
#pragma unroll
      for (int n0 = 0; n0 < 32; n0 += 8) {
        v8bf r0 = *(const v8bf*)(vr0 + n0);
        v8bf r1 = *(const v8bf*)(vr1 + n0);
#pragma unroll
        for (int i = 0; i < 8; ++i) {
          v2bf pk; pk[0] = r0[i]; pk[1] = r1[i];
          *(v2bf*)&Vt[(fsel + n0 + i) * 32 + 2 * t] = pk;
        }
      }
    }

    // ---- scores: S = Q @ K^T for two 16-key subtiles ----
    v8f st[2];
#pragma unroll
    for (int sub = 0; sub < 2; ++sub) {
      int jb = j0 + sub * 16;
      v8f c;
#pragma unroll
      for (int r = 0; r < 8; ++r) c[r] = 0.0f;
#pragma unroll
      for (int kc = 0; kc < 2; ++kc) {
        // B[k][n] = K[jb+n][kc*32 + k] -> per-lane contiguous row segment
        const bf16* kp =
            Kb + ((size_t)b * S_ + jb + lm) * DO_ + kc * 32 + (half ? 16 : 0);
        v16bf bm = *(const v16bf*)kp;
        c = wmma_bf16(qa[kc], bm, c);
      }
#pragma unroll
      for (int r = 0; r < 8; ++r) {
        int i = qbase + r + 8 * half;
        int j = jb + lm;
        float sv = c[r] * 0.125f;               // 1/sqrt(64)
        c[r] = (j > i) ? -__builtin_inff() : sv;
      }
      st[sub] = c;
    }

    // ---- online softmax update ----
    float alpha[8];
#pragma unroll
    for (int r = 0; r < 8; ++r) {
      float mx = fmaxf(st[0][r], st[1][r]);
#pragma unroll
      for (int d = 1; d < 16; d <<= 1) mx = fmaxf(mx, __shfl_xor(mx, d, 32));
      float nmax = fmaxf(M_[r], mx);            // finite: j0 <= i always holds
      alpha[r] = __expf(M_[r] - nmax);
      M_[r] = nmax;
    }
#pragma unroll
    for (int sub = 0; sub < 2; ++sub)
#pragma unroll
      for (int r = 0; r < 8; ++r) st[sub][r] = __expf(st[sub][r] - M_[r]);
#pragma unroll
    for (int r = 0; r < 8; ++r) {
      float rs = st[0][r] + st[1][r];
#pragma unroll
      for (int d = 1; d < 16; d <<= 1) rs += __shfl_xor(rs, d, 32);
      L_[r] = L_[r] * alpha[r] + rs;
    }
#pragma unroll
    for (int c = 0; c < 4; ++c)
#pragma unroll
      for (int r = 0; r < 8; ++r) O[c][r] *= alpha[r];

    // ---- P (D-layout) -> LDS -> reload in A-layout ----
#pragma unroll
    for (int sub = 0; sub < 2; ++sub)
#pragma unroll
      for (int r = 0; r < 8; ++r)
        Pt[(r + 8 * half) * 32 + sub * 16 + lm] = (bf16)st[sub][r];
    __syncthreads();

    v16bf pa;
    {
      const bf16* pr = Pt + lm * 32 + abase;
      v8bf lo = *(const v8bf*)(pr);
      v8bf hi = *(const v8bf*)(pr + 8);
#pragma unroll
      for (int i = 0; i < 8; ++i) { pa[i] = lo[i]; pa[8 + i] = hi[i]; }
    }

    // ---- O += P @ Vtile (4 feature chunks of 16) ----
#pragma unroll
    for (int c = 0; c < 4; ++c) {
      const bf16* vp = Vt + (c * 16 + lm) * 32 + (half ? 16 : 0);
      v16bf bm = *(const v16bf*)vp;
      O[c] = wmma_bf16(pa, bm, O[c]);
    }
  }

  // ---- epilogue: normalize and store f32 output ----
#pragma unroll
  for (int r = 0; r < 8; ++r) {
    float inv = 1.0f / L_[r];
    int i = qbase + r + 8 * half;
#pragma unroll
    for (int c = 0; c < 4; ++c)
      out[((size_t)b * S_ + i) * DO_ + c * 16 + lm] = O[c][r] * inv;
  }
}

// ---------------------------------------------------------------------------
extern "C" void kernel_launch(void* const* d_in, const int* in_sizes, int n_in,
                              void* d_out, int out_size, void* d_ws, size_t ws_size,
                              hipStream_t stream) {
  const float* q  = (const float*)d_in[0];
  const float* k  = (const float*)d_in[1];
  const float* v  = (const float*)d_in[2];
  // d_in[3] = causal mask -> computed analytically in-kernel
  const float* Wq = (const float*)d_in[4];
  const float* bq = (const float*)d_in[5];
  const float* Wk = (const float*)d_in[6];
  const float* bk = (const float*)d_in[7];
  const float* Wv = (const float*)d_in[8];
  const float* bv = (const float*)d_in[9];

  bf16* wt   = (bf16*)d_ws;                                // 384 KB
  bf16* proj = (bf16*)((char*)d_ws + WT_BYTES);            // 3 x 2 MB bf16 Q/K/V
  float* out = (float*)d_out;

  int ttot = 3 * DO_ * DM_;
  wt_transpose_kernel<<<(ttot + 255) / 256, 256, 0, stream>>>(Wq, Wk, Wv, wt);

  proj_kernel<<<dim3((B_ * S_) / 16, 3), 32, 0, stream>>>(q, k, v, bq, bk, bv,
                                                          wt, proj);

  attn_kernel<<<dim3(S_ / 16, B_), 32, 0, stream>>>(proj, out);
}
